// MyGCNlayer_44547400794683
// MI455X (gfx1250) — compile-verified, matrix operand order checked by reading, not run on hardware
//
#include <hip/hip_runtime.h>
#include <hip/hip_bf16.h>

typedef __attribute__((ext_vector_type(16))) __bf16 v16bf;
typedef __attribute__((ext_vector_type(8)))  float  v8f;

#define IN_DIM  256
#define OUT_DIM 256

// ---------------------------------------------------------------------------
// Phase 1: edge scatter  agg[dst[e]] += feature[src[e]]
// One wave (32 lanes) per edge; each lane handles 8 contiguous floats.
// ---------------------------------------------------------------------------
__global__ __launch_bounds__(256) void gcn_scatter_kernel(
    const float* __restrict__ feature,
    const int*   __restrict__ src,
    const int*   __restrict__ dst,
    float*       __restrict__ agg,
    int nEdges)
{
    unsigned tid  = blockIdx.x * 256u + threadIdx.x;
    int      edge = (int)(tid >> 5);
    if (edge >= nEdges) return;
    int lane = (int)(tid & 31u);

    int s = src[edge];
    int d = dst[edge];

    const float4* fs = (const float4*)(feature + (size_t)s * IN_DIM) + lane * 2;
    float4 x0 = fs[0];
    float4 x1 = fs[1];

    float* ad = agg + (size_t)d * IN_DIM + lane * 8;
    unsafeAtomicAdd(ad + 0, x0.x);
    unsafeAtomicAdd(ad + 1, x0.y);
    unsafeAtomicAdd(ad + 2, x0.z);
    unsafeAtomicAdd(ad + 3, x0.w);
    unsafeAtomicAdd(ad + 4, x1.x);
    unsafeAtomicAdd(ad + 5, x1.y);
    unsafeAtomicAdd(ad + 6, x1.z);
    unsafeAtomicAdd(ad + 7, x1.w);
}

// ---------------------------------------------------------------------------
// Phase 2: out = agg @ W via v_wmma_f32_16x16x32_bf16 with split-bf16
// (hi+lo) operands for near-f32 accuracy:  C += Ahi*Bhi + Ahi*Blo + Alo*Bhi.
// Block: 8 waves, each wave owns one 16x16 output tile; blockIdx.y = N-tile.
// W column-slab staged in LDS as packed bf16-pair dwords (K-major).
// ---------------------------------------------------------------------------
__global__ __launch_bounds__(256) void gcn_wmma_gemm_kernel(
    const float* __restrict__ agg,
    const float* __restrict__ weight,
    float*       __restrict__ out,
    int nNodes)
{
    // LDS: [pair p = k/2][col c]  packed (bf16(k=2p) | bf16(k=2p+1)<<16)
    __shared__ unsigned sbhi[(IN_DIM / 2) * 16];
    __shared__ unsigned sblo[(IN_DIM / 2) * 16];

    const int ntile = blockIdx.y;
    const int ncol0 = ntile * 16;

    // ---- stage W slab (256 K x 16 N) as bf16 hi/lo packed pairs ----
    for (int idx = threadIdx.x; idx < (IN_DIM / 2) * 16; idx += 256) {
        int p = idx >> 4;       // k-pair index
        int c = idx & 15;       // column within tile
        int k = p * 2;
        float w0 = weight[(size_t)k       * OUT_DIM + ncol0 + c];
        float w1 = weight[(size_t)(k + 1) * OUT_DIM + ncol0 + c];
        __bf16 h0 = (__bf16)w0, h1 = (__bf16)w1;
        __bf16 l0 = (__bf16)(w0 - (float)h0);
        __bf16 l1 = (__bf16)(w1 - (float)h1);
        union { __bf16 h[2]; unsigned u; } pk;
        pk.h[0] = h0; pk.h[1] = h1; sbhi[idx] = pk.u;
        pk.h[0] = l0; pk.h[1] = l1; sblo[idx] = pk.u;
    }
    __syncthreads();

    const int wave  = threadIdx.x >> 5;
    const int lane  = threadIdx.x & 31;
    const int mtile = blockIdx.x * 8 + wave;
    if (mtile * 16 >= nNodes) return;      // whole-wave uniform exit

    const int col   = lane & 15;           // B column / D column (N)
    const int kbase = (lane >> 4) * 8;     // K split between lane halves

    int arow = mtile * 16 + (lane & 15);   // A row (M)
    if (arow >= nNodes) arow = nNodes - 1; // clamp (stores are guarded)
    const float* aptr = agg + (size_t)arow * IN_DIM;

    v8f acc = {};

#pragma unroll
    for (int t = 0; t < IN_DIM / 32; ++t) {
        const int k0 = t * 32;

        // ---- A operand: 16 floats of this row -> bf16 hi/lo ----
        float av[16];
        float4 t0 = *(const float4*)(aptr + k0 + kbase);
        float4 t1 = *(const float4*)(aptr + k0 + kbase + 4);
        float4 t2 = *(const float4*)(aptr + k0 + 16 + kbase);
        float4 t3 = *(const float4*)(aptr + k0 + 16 + kbase + 4);
        av[0] = t0.x; av[1] = t0.y; av[2]  = t0.z; av[3]  = t0.w;
        av[4] = t1.x; av[5] = t1.y; av[6]  = t1.z; av[7]  = t1.w;
        av[8] = t2.x; av[9] = t2.y; av[10] = t2.z; av[11] = t2.w;
        av[12] = t3.x; av[13] = t3.y; av[14] = t3.z; av[15] = t3.w;

        union { __bf16 h[16]; v16bf v; } ahi, alo;
#pragma unroll
        for (int i = 0; i < 16; ++i) {
            __bf16 h = (__bf16)av[i];
            ahi.h[i] = h;
            alo.h[i] = (__bf16)(av[i] - (float)h);
        }

        // ---- B operand: packed pairs from LDS ----
        union { unsigned u[8]; v16bf v; } bhi, blo;
        const int p0 = (k0 + kbase) >> 1;
        const int p1 = (k0 + 16 + kbase) >> 1;
#pragma unroll
        for (int j = 0; j < 4; ++j) {
            bhi.u[j]     = sbhi[(p0 + j) * 16 + col];
            blo.u[j]     = sblo[(p0 + j) * 16 + col];
            bhi.u[4 + j] = sbhi[(p1 + j) * 16 + col];
            blo.u[4 + j] = sblo[(p1 + j) * 16 + col];
        }

        // ---- split-precision accumulate: 3 WMMAs ----
        acc = __builtin_amdgcn_wmma_f32_16x16x32_bf16(
            false, ahi.v, false, bhi.v, (short)0, acc, false, false);
        acc = __builtin_amdgcn_wmma_f32_16x16x32_bf16(
            false, ahi.v, false, blo.v, (short)0, acc, false, false);
        acc = __builtin_amdgcn_wmma_f32_16x16x32_bf16(
            false, alo.v, false, bhi.v, (short)0, acc, false, false);
    }

    // ---- store D: VGPR j -> M = j + 8*(lane>=16), N = lane&15 ----
    const int rbase = mtile * 16 + ((lane >> 4) << 3);
    const int ncol  = ncol0 + col;
#pragma unroll
    for (int j = 0; j < 8; ++j) {
        int r = rbase + j;
        if (r < nNodes) out[(size_t)r * OUT_DIM + ncol] = acc[j];
    }
}

// ---------------------------------------------------------------------------
extern "C" void kernel_launch(void* const* d_in, const int* in_sizes, int n_in,
                              void* d_out, int out_size, void* d_ws, size_t ws_size,
                              hipStream_t stream)
{
    const float* feature = (const float*)d_in[0];
    const float* weight  = (const float*)d_in[1];
    const int*   src     = (const int*)d_in[2];
    const int*   dst     = (const int*)d_in[3];
    float*       out     = (float*)d_out;
    float*       agg     = (float*)d_ws;

    const int nNodes = in_sizes[0] / IN_DIM;
    const int nEdges = in_sizes[2];

    // zero the aggregation buffer (capture-safe async memset)
    hipMemsetAsync(agg, 0, (size_t)nNodes * IN_DIM * sizeof(float), stream);

    // Phase 1: scatter-add (one wave per edge)
    {
        long long threads = (long long)nEdges * 32;
        int blocks = (int)((threads + 255) / 256);
        gcn_scatter_kernel<<<blocks, 256, 0, stream>>>(feature, src, dst, agg, nEdges);
    }

    // Phase 2: WMMA GEMM
    {
        int mtiles = (nNodes + 15) / 16;
        dim3 grid((mtiles + 7) / 8, OUT_DIM / 16);
        gcn_wmma_gemm_kernel<<<grid, 256, 0, stream>>>(agg, weight, out, nNodes);
    }
}
